// Seq2Seq_64338610094322
// MI455X (gfx1250) — matrix operand order repaired and checked by reading, not compiled, not run on hardware
//
#include <hip/hip_runtime.h>
#include <hip/hip_bf16.h>

// Seq2Seq beam search for MI455X (gfx1250, wave32).
// Heavy GEMM ([4,2048]x[2048,32000]) runs as bf16 WMMA against a one-time
// bf16-transposed copy of W_out (131MB -> resident in the 192MB L2).
// GRU/attention matvecs stay fp32 VALU against transposed weights (coalesced).
// Workspace requirement: ~205 MB.

#define H      1024
#define H2     2048
#define H3     3072
#define VOCAB  32000
#define KBEAM  4
#define TMAX   64
#define SLEN   512
#define EOS_TOK 2
#define PAD_TOK 0
#define NEGINF (-1e9f)

typedef __attribute__((ext_vector_type(16))) __bf16 v16bf;
typedef __attribute__((ext_vector_type(8)))  __bf16 v8bf;
typedef __attribute__((ext_vector_type(8)))  float  v8f;

__device__ __forceinline__ float wred_sum(float v) {
#pragma unroll
  for (int o = 16; o; o >>= 1) v += __shfl_xor(v, o, 32);
  return v;
}
__device__ __forceinline__ float wred_max(float v) {
#pragma unroll
  for (int o = 16; o; o >>= 1) v = fmaxf(v, __shfl_xor(v, o, 32));
  return v;
}
__device__ __forceinline__ float sigm(float x) { return 1.f / (1.f + __expf(-x)); }

// ---------------- one-time weight preprocessing ----------------

// out[c*R + r] = in[r*C + c]   (in: [R][C] f32 -> out: [C][R] f32)
__global__ void transpose_f32_kernel(const float* __restrict__ in, float* __restrict__ out,
                                     int R, int C) {
  __shared__ float tile[32][33];
  int c0 = blockIdx.x * 32, r0 = blockIdx.y * 32;
  for (int j = threadIdx.y; j < 32; j += 8) {
    int r = r0 + j, c = c0 + threadIdx.x;
    tile[j][threadIdx.x] = (r < R && c < C) ? in[(size_t)r * C + c] : 0.f;
  }
  __syncthreads();
  for (int j = threadIdx.y; j < 32; j += 8) {
    int c = c0 + j, r = r0 + threadIdx.x;
    if (c < C && r < R) out[(size_t)c * R + r] = tile[threadIdx.x][j];
  }
}

// out[c*R + r] = bf16(in[r*C + c])  (W_out: [2048][32000] f32 -> [32000][2048] bf16)
__global__ void transpose_f32_bf16_kernel(const float* __restrict__ in, __bf16* __restrict__ out,
                                          int R, int C) {
  __shared__ float tile[32][33];
  int c0 = blockIdx.x * 32, r0 = blockIdx.y * 32;
  for (int j = threadIdx.y; j < 32; j += 8) {
    int r = r0 + j, c = c0 + threadIdx.x;
    tile[j][threadIdx.x] = (r < R && c < C) ? in[(size_t)r * C + c] : 0.f;
  }
  __syncthreads();
  for (int j = threadIdx.y; j < 32; j += 8) {
    int c = c0 + j, r = r0 + threadIdx.x;
    if (c < C && r < R) out[(size_t)c * R + r] = (__bf16)tile[threadIdx.x][j];
  }
}

__global__ void init_kernel(const int* __restrict__ trg, int* tokens, float* henc,
                            __bf16* xo) {
  int tid = blockIdx.x * blockDim.x + threadIdx.x;
  if (tid == 0) tokens[0] = trg[0];          // greedy seed token = trg[0,0]
  if (tid < H) henc[tid] = 0.f;              // encoder h0 = 0
  for (int i = tid; i < 16 * H2; i += gridDim.x * blockDim.x) xo[i] = (__bf16)0.f;
}

// ---------------- encoder ----------------

// one wave per hidden index i: 6 coalesced 1024-dots + GRU cell update
__global__ __launch_bounds__(256) void enc_step_kernel(
    const int* __restrict__ src, const float* __restrict__ emb_src,
    const float* __restrict__ WxeT, const float* __restrict__ WheT,
    const float* __restrict__ b_e,
    const float* __restrict__ h_in, float* __restrict__ h_out,
    float* __restrict__ enc_out_row, int t) {
  const int lane = threadIdx.x & 31;
  const int i = (blockIdx.x * blockDim.x + threadIdx.x) >> 5;   // 0..1023
  const int tok = src[t];
  const float* __restrict__ x = emb_src + (size_t)tok * H;
  const float* wz = WxeT + (size_t)i * H;
  const float* wr = WxeT + (size_t)(i + H) * H;
  const float* wn = WxeT + (size_t)(i + 2 * H) * H;
  const float* uz = WheT + (size_t)i * H;
  const float* ur = WheT + (size_t)(i + H) * H;
  const float* un = WheT + (size_t)(i + 2 * H) * H;
  float gz = 0, gr = 0, gn = 0, hz = 0, hr = 0, hn = 0;
#pragma unroll 2
  for (int p = 0; p < 8; p++) {
    int c = p * 128 + lane * 4;
    float4 xv = *(const float4*)(x + c);
    float4 hv = *(const float4*)(h_in + c);
    float4 w;
    w = *(const float4*)(wz + c); gz += xv.x * w.x + xv.y * w.y + xv.z * w.z + xv.w * w.w;
    w = *(const float4*)(wr + c); gr += xv.x * w.x + xv.y * w.y + xv.z * w.z + xv.w * w.w;
    w = *(const float4*)(wn + c); gn += xv.x * w.x + xv.y * w.y + xv.z * w.z + xv.w * w.w;
    w = *(const float4*)(uz + c); hz += hv.x * w.x + hv.y * w.y + hv.z * w.z + hv.w * w.w;
    w = *(const float4*)(ur + c); hr += hv.x * w.x + hv.y * w.y + hv.z * w.z + hv.w * w.w;
    w = *(const float4*)(un + c); hn += hv.x * w.x + hv.y * w.y + hv.z * w.z + hv.w * w.w;
  }
  gz = wred_sum(gz); gr = wred_sum(gr); gn = wred_sum(gn);
  hz = wred_sum(hz); hr = wred_sum(hr); hn = wred_sum(hn);
  if (lane == 0) {
    float z = sigm(gz + b_e[i] + hz);
    float r = sigm(gr + b_e[i + H] + hr);
    float n = tanhf(gn + b_e[i + 2 * H] + r * hn);
    float ho = (1.f - z) * n + z * h_in[i];
    h_out[i] = ho;
    enc_out_row[i] = ho;
  }
}

// out[i] = tanh(dot(in, WT[i]) + b[i]); one wave per i
__global__ __launch_bounds__(256) void linear_tanh_kernel(
    const float* __restrict__ WT, const float* __restrict__ b,
    const float* __restrict__ in, float* __restrict__ out) {
  const int lane = threadIdx.x & 31;
  const int i = (blockIdx.x * blockDim.x + threadIdx.x) >> 5;
  const float* w = WT + (size_t)i * H;
  float acc = 0.f;
#pragma unroll 2
  for (int p = 0; p < 8; p++) {
    int c = p * 128 + lane * 4;
    float4 xv = *(const float4*)(in + c);
    float4 wv = *(const float4*)(w + c);
    acc += xv.x * wv.x + xv.y * wv.y + xv.z * wv.z + xv.w * wv.w;
  }
  acc = wred_sum(acc);
  if (lane == 0) out[i] = tanhf(acc + b[i]);
}

// ---------------- decoder: attention + context + gather emb ----------------
// one block (1024 thr) per beam; scores+softmax stay in LDS; writes xg=[emb|ctx]
__global__ __launch_bounds__(1024) void dec_attn_kernel(
    const float* __restrict__ enc_out, const float* __restrict__ hid,
    const int* __restrict__ tokens, const float* __restrict__ emb_tgt,
    float* __restrict__ xg) {
  __shared__ float sh[H];
  __shared__ float sc[SLEN];
  __shared__ float red[32];
  const int tid = threadIdx.x, lane = tid & 31, wv = tid >> 5;
  const int k = blockIdx.x;
  sh[tid] = hid[(size_t)k * H + tid];
  __syncthreads();
  for (int q = 0; q < 16; q++) {          // 32 waves x 16 = 512 scores
    int s = wv * 16 + q;
    const float* e = enc_out + (size_t)s * H;
    float a = 0.f;
#pragma unroll 4
    for (int p = 0; p < 32; p++) { int c = p * 32 + lane; a += sh[c] * e[c]; }
    a = wred_sum(a);
    if (lane == 0) sc[s] = a;
  }
  __syncthreads();
  // softmax over SLEN
  float v = (tid < SLEN) ? sc[tid] : -3.0e38f;
  v = wred_max(v);
  if (lane == 0) red[wv] = v;
  __syncthreads();
  if (tid < 32) { float r = red[tid]; r = wred_max(r); if (tid == 0) red[0] = r; }
  __syncthreads();
  float mx = red[0];
  __syncthreads();
  float e = (tid < SLEN) ? __expf(sc[tid] - mx) : 0.f;
  float s = wred_sum(e);
  if (lane == 0) red[wv] = s;
  __syncthreads();
  if (tid < 32) { float r = red[tid]; r = wred_sum(r); if (tid == 0) red[0] = r; }
  __syncthreads();
  float inv = 1.f / red[0];
  if (tid < SLEN) sc[tid] = e * inv;
  __syncthreads();
  // emb gather + ctx = attn @ enc_out
  int tok = tokens[k];
  xg[(size_t)k * H2 + tid] = emb_tgt[(size_t)tok * H + tid];
  float acc = 0.f;
#pragma unroll 4
  for (int sI = 0; sI < SLEN; sI++) acc += sc[sI] * enc_out[(size_t)sI * H + tid];
  xg[(size_t)k * H2 + H + tid] = acc;
}

// ---------------- decoder GRU cell; also packs bf16 A = [h_new | ctx] ----------------
__global__ __launch_bounds__(256) void dec_gru_kernel(
    const float* __restrict__ WxdT, const float* __restrict__ WhdT,
    const float* __restrict__ b_d,
    const float* __restrict__ xg, const float* __restrict__ hid_in,
    float* __restrict__ hid_out, __bf16* __restrict__ xo) {
  const int lane = threadIdx.x & 31;
  const int wid = (blockIdx.x * blockDim.x + threadIdx.x) >> 5;
  const int k = wid >> 10;
  const int i = wid & (H - 1);
  const float* x = xg + (size_t)k * H2;
  const float* h = hid_in + (size_t)k * H;
  const float* wz = WxdT + (size_t)i * H2;
  const float* wr = WxdT + (size_t)(i + H) * H2;
  const float* wn = WxdT + (size_t)(i + 2 * H) * H2;
  const float* uz = WhdT + (size_t)i * H;
  const float* ur = WhdT + (size_t)(i + H) * H;
  const float* un = WhdT + (size_t)(i + 2 * H) * H;
  float gz = 0, gr = 0, gn = 0, hz = 0, hr = 0, hn = 0;
  for (int p = 0; p < 16; p++) {
    int c = p * 128 + lane * 4;
    float4 xv = *(const float4*)(x + c);
    float4 w;
    w = *(const float4*)(wz + c); gz += xv.x * w.x + xv.y * w.y + xv.z * w.z + xv.w * w.w;
    w = *(const float4*)(wr + c); gr += xv.x * w.x + xv.y * w.y + xv.z * w.z + xv.w * w.w;
    w = *(const float4*)(wn + c); gn += xv.x * w.x + xv.y * w.y + xv.z * w.z + xv.w * w.w;
    if (p < 8) {
      float4 hv = *(const float4*)(h + c);
      w = *(const float4*)(uz + c); hz += hv.x * w.x + hv.y * w.y + hv.z * w.z + hv.w * w.w;
      w = *(const float4*)(ur + c); hr += hv.x * w.x + hv.y * w.y + hv.z * w.z + hv.w * w.w;
      w = *(const float4*)(un + c); hn += hv.x * w.x + hv.y * w.y + hv.z * w.z + hv.w * w.w;
    }
  }
  gz = wred_sum(gz); gr = wred_sum(gr); gn = wred_sum(gn);
  hz = wred_sum(hz); hr = wred_sum(hr); hn = wred_sum(hn);
  if (lane == 0) {
    float z = sigm(gz + b_d[i] + hz);
    float r = sigm(gr + b_d[i + H] + hr);
    float n = tanhf(gn + b_d[i + 2 * H] + r * hn);
    float ho = (1.f - z) * n + z * h[i];
    hid_out[(size_t)k * H + i] = ho;
    xo[(size_t)k * H2 + i] = (__bf16)ho;                 // logits A row: [h | ctx]
    xo[(size_t)k * H2 + H + i] = (__bf16)x[H + i];
  }
}

// ---------------- logits GEMM: [16,2048](bf16) x [2048,32000](bf16,T) via WMMA ----------------
// one wave per 16-column tile (2000 tiles); 64 k-chunks of 32.
__global__ __launch_bounds__(256) void dec_logits_kernel(
    const __bf16* __restrict__ xo, const __bf16* __restrict__ WoutT,
    const float* __restrict__ b_out, float* __restrict__ logits) {
  const int lane = threadIdx.x & 31;
  const int wv = threadIdx.x >> 5;
  const int tile = blockIdx.x * 8 + wv;
  const int col0 = tile * 16;
  const int half = lane >> 4;
  const int l16 = lane & 15;
  const __bf16* ap = xo + (size_t)l16 * H2;                        // A row m=l16
  const __bf16* bp = WoutT + (size_t)(col0 + l16) * H2 + half * 16; // B col, K-contig
  v8f acc = {0.f, 0.f, 0.f, 0.f, 0.f, 0.f, 0.f, 0.f};
  for (int kc = 0; kc < H2; kc += 32) {
    // A 16x32 bf16 layout (matches B): half0 -> K{0..15}, half1 -> K{16..31}
    v8bf alo = *(const v8bf*)(ap + kc + half * 16);
    v8bf ahi = *(const v8bf*)(ap + kc + half * 16 + 8);
    v16bf a;
#pragma unroll
    for (int j = 0; j < 8; j++) { a[j] = alo[j]; a[8 + j] = ahi[j]; }
    // B 32x16 bf16: lane half selects K 0..15 / 16..31, contiguous after transpose
    v16bf b = *(const v16bf*)(bp + kc);
    __builtin_prefetch(bp + kc + 32, 0, 1);
    acc = __builtin_amdgcn_wmma_f32_16x16x32_bf16(
        false, a, false, b, (short)0, acc, false, false);
  }
  if (lane < 16) {
#pragma unroll
    for (int r = 0; r < KBEAM; r++) {   // D: VGPR r holds row M=r on lanes 0..15
      logits[(size_t)r * VOCAB + col0 + lane] = acc[r] + b_out[col0 + lane];
    }
  }
}

// ---------------- greedy seed: argmax over V, broadcast beam state ----------------
__global__ __launch_bounds__(1024) void greedy_pick_kernel(
    const float* __restrict__ logits, int* tokens, float* logp, int* lens, int* fin,
    int* seqs, const float* __restrict__ hidB, float* __restrict__ hidA) {
  __shared__ float rv[32];
  __shared__ int ri[32];
  __shared__ int s_tok;
  const int tid = threadIdx.x, lane = tid & 31, wv = tid >> 5;
  float bv = -3.4e38f; int bix = 0;
  for (int c = tid; c < VOCAB; c += 1024) {
    float xv = logits[c];
    if (xv > bv) { bv = xv; bix = c; }
  }
#pragma unroll
  for (int off = 16; off; off >>= 1) {
    float ov = __shfl_xor(bv, off, 32); int oi = __shfl_xor(bix, off, 32);
    if (ov > bv || (ov == bv && oi < bix)) { bv = ov; bix = oi; }
  }
  if (lane == 0) { rv[wv] = bv; ri[wv] = bix; }
  __syncthreads();
  if (tid == 0) {
    float v = rv[0]; int ix = ri[0];
    for (int w = 1; w < 32; w++)
      if (rv[w] > v || (rv[w] == v && ri[w] < ix)) { v = rv[w]; ix = ri[w]; }
    s_tok = ix;
  }
  __syncthreads();
  int ft = s_tok;
  if (tid < KBEAM) { tokens[tid] = ft; logp[tid] = 0.f; lens[tid] = 1; fin[tid] = 0; }
  if (tid < KBEAM * TMAX) seqs[tid] = ((tid & (TMAX - 1)) == 0) ? ft : PAD_TOK;
#pragma unroll
  for (int j = 0; j < KBEAM; j++) hidA[(size_t)j * H + tid] = hidB[tid];
}

__device__ __forceinline__ void top4_insert(float sc, float cd, int ix,
                                            float* bs, float* bc, int* bi) {
  if (sc > bs[3]) {
    int p = 3;
    while (p > 0 && sc > bs[p - 1]) {
      bs[p] = bs[p - 1]; bc[p] = bc[p - 1]; bi[p] = bi[p - 1]; p--;
    }
    bs[p] = sc; bc[p] = cd; bi[p] = ix;
  }
}

// ---------------- beam step: log_softmax + length-norm score + top-4 + reorder ----------------
__global__ __launch_bounds__(1024) void beam_topk_kernel(
    const float* __restrict__ logits,
    int* tokens, float* logp, int* lens, int* fin, int* seqs,
    const float* __restrict__ hidB, float* __restrict__ hidA, int t) {
  __shared__ float red[32];
  __shared__ float s_lse[KBEAM];
  __shared__ float s_oldlogp[KBEAM];
  __shared__ int s_oldlen[KBEAM], s_oldfin[KBEAM];
  __shared__ int s_seq[KBEAM * TMAX];
  __shared__ float s_ms[128]; __shared__ float s_mc[128]; __shared__ int s_mi[128];
  __shared__ int s_bsel[KBEAM], s_tok[KBEAM], s_pfin[KBEAM];
  const int tid = threadIdx.x, lane = tid & 31, wv = tid >> 5;
  if (tid < KBEAM) { s_oldlogp[tid] = logp[tid]; s_oldlen[tid] = lens[tid]; s_oldfin[tid] = fin[tid]; }
  if (tid < KBEAM * TMAX) s_seq[tid] = seqs[tid];
  __syncthreads();

  for (int k = 0; k < KBEAM; k++) {     // logsumexp per beam
    float m = -3.0e38f;
    for (int c = tid; c < VOCAB; c += 1024) m = fmaxf(m, logits[(size_t)k * VOCAB + c]);
    m = wred_max(m);
    if (lane == 0) red[wv] = m;
    __syncthreads();
    if (tid < 32) { float r = red[tid]; r = wred_max(r); if (tid == 0) red[0] = r; }
    __syncthreads();
    float mx = red[0];
    __syncthreads();
    float s = 0.f;
    for (int c = tid; c < VOCAB; c += 1024) s += __expf(logits[(size_t)k * VOCAB + c] - mx);
    s = wred_sum(s);
    if (lane == 0) red[wv] = s;
    __syncthreads();
    if (tid < 32) { float r = red[tid]; r = wred_sum(r); if (tid == 0) red[0] = r; }
    __syncthreads();
    if (tid == 0) s_lse[k] = mx + __logf(red[0]);
    __syncthreads();
  }

  float bs[4] = {-3.4e38f, -3.4e38f, -3.4e38f, -3.4e38f};
  float bc[4] = {NEGINF, NEGINF, NEGINF, NEGINF};
  int bi[4] = {0, 0, 0, 0};
  for (int idx = tid; idx < KBEAM * VOCAB; idx += 1024) {
    int k = idx / VOCAB;
    int c = idx - k * VOCAB;
    float ll = logits[idx] - s_lse[k];
    int f = s_oldfin[k];
    float cand = s_oldlogp[k] + (f ? ((c == PAD_TOK) ? 0.f : NEGINF) : ll);
    int nl = f ? s_oldlen[k] : s_oldlen[k] + 1;
    float sc = cand / ((float)(nl - 1) + 1e-6f);
    if (t == 0 && k > 0) sc = NEGINF;
    top4_insert(sc, cand, idx, bs, bc, bi);
  }
  // wave-level merge of sorted top-4 lists
  for (int off = 16; off; off >>= 1) {
    float os[4], oc[4]; int oi[4];
#pragma unroll
    for (int j = 0; j < 4; j++) {
      os[j] = __shfl_xor(bs[j], off, 32);
      oc[j] = __shfl_xor(bc[j], off, 32);
      oi[j] = __shfl_xor(bi[j], off, 32);
    }
    float ns[4], nc[4]; int ni[4];
    int a = 0, b = 0;
#pragma unroll
    for (int j = 0; j < 4; j++) {
      bool ta = bs[a] >= os[b];
      ns[j] = ta ? bs[a] : os[b];
      nc[j] = ta ? bc[a] : oc[b];
      ni[j] = ta ? bi[a] : oi[b];
      if (ta) a++; else b++;
    }
#pragma unroll
    for (int j = 0; j < 4; j++) { bs[j] = ns[j]; bc[j] = nc[j]; bi[j] = ni[j]; }
  }
  if (lane == 0) {
#pragma unroll
    for (int j = 0; j < 4; j++) { s_ms[wv * 4 + j] = bs[j]; s_mc[wv * 4 + j] = bc[j]; s_mi[wv * 4 + j] = bi[j]; }
  }
  __syncthreads();
  if (tid == 0) {
    float fs[4] = {-3.4e38f, -3.4e38f, -3.4e38f, -3.4e38f};
    float fc[4] = {NEGINF, NEGINF, NEGINF, NEGINF};
    int fi[4] = {0, 0, 0, 0};
    for (int e = 0; e < 128; e++) top4_insert(s_ms[e], s_mc[e], s_mi[e], fs, fc, fi);
    for (int j = 0; j < KBEAM; j++) {
      int bk = fi[j] / VOCAB;
      int tk = fi[j] - bk * VOCAB;
      int pf = s_oldfin[bk];
      int nl = pf ? s_oldlen[bk] : s_oldlen[bk] + 1;
      int nf = (pf || (tk == EOS_TOK)) ? 1 : 0;
      tokens[j] = tk; logp[j] = fc[j]; lens[j] = nl; fin[j] = nf;
      s_bsel[j] = bk; s_tok[j] = tk; s_pfin[j] = pf;
    }
  }
  __syncthreads();
  if (tid < KBEAM * TMAX) {
    int j = tid >> 6, p = tid & (TMAX - 1);
    int v = s_seq[s_bsel[j] * TMAX + p];
    if (p == t + 1) v = s_pfin[j] ? PAD_TOK : s_tok[j];
    seqs[tid] = v;
  }
#pragma unroll
  for (int j = 0; j < KBEAM; j++)
    hidA[(size_t)j * H + tid] = hidB[(size_t)s_bsel[j] * H + tid];
}

__global__ void finalize_kernel(const int* __restrict__ seqs, const float* __restrict__ logp,
                                const int* __restrict__ lens, float* __restrict__ out) {
  int tid = blockIdx.x * blockDim.x + threadIdx.x;
  if (tid < KBEAM * TMAX) out[tid] = (float)seqs[tid];
  else if (tid < KBEAM * TMAX + KBEAM) {
    int j = tid - KBEAM * TMAX;
    out[tid] = logp[j] / ((float)(lens[j] - 1) + 1e-6f);
  }
}

// ---------------- host orchestration ----------------

extern "C" void kernel_launch(void* const* d_in, const int* in_sizes, int n_in,
                              void* d_out, int out_size, void* d_ws, size_t ws_size,
                              hipStream_t stream) {
  (void)in_sizes; (void)n_in; (void)out_size; (void)ws_size;
  const int*   src     = (const int*)d_in[0];
  const int*   trg     = (const int*)d_in[1];
  const float* emb_src = (const float*)d_in[2];
  const float* emb_tgt = (const float*)d_in[3];
  const float* Wx_e    = (const float*)d_in[4];
  const float* Wh_e    = (const float*)d_in[5];
  const float* b_e     = (const float*)d_in[6];
  const float* W_encl  = (const float*)d_in[7];
  const float* b_encl  = (const float*)d_in[8];
  const float* W_lin   = (const float*)d_in[9];
  const float* b_lin   = (const float*)d_in[10];
  const float* Wx_d    = (const float*)d_in[11];
  const float* Wh_d    = (const float*)d_in[12];
  const float* b_d     = (const float*)d_in[13];
  const float* W_out   = (const float*)d_in[14];
  const float* b_out   = (const float*)d_in[15];

  char* ws = (char*)d_ws;
  size_t off = 0;
  auto take = [&](size_t bytes) -> char* {
    char* p = ws + off;
    off = (off + bytes + 255) & ~(size_t)255;
    return p;
  };
  float*  WxeT   = (float*)take((size_t)H3 * H * 4);
  float*  WheT   = (float*)take((size_t)H3 * H * 4);
  float*  WxdT   = (float*)take((size_t)H3 * H2 * 4);
  float*  WhdT   = (float*)take((size_t)H3 * H * 4);
  float*  WenclT = (float*)take((size_t)H * H * 4);
  float*  WlinT  = (float*)take((size_t)H * H * 4);
  __bf16* WoutT  = (__bf16*)take((size_t)VOCAB * H2 * 2);   // 131 MB, L2-resident
  float*  enc_out = (float*)take((size_t)SLEN * H * 4);
  float*  henc_a = (float*)take(H * 4);
  float*  henc_b = (float*)take(H * 4);
  float*  hidA   = (float*)take((size_t)KBEAM * H * 4);
  float*  hidB   = (float*)take((size_t)KBEAM * H * 4);
  float*  xg     = (float*)take((size_t)KBEAM * H2 * 4);
  __bf16* xo     = (__bf16*)take((size_t)16 * H2 * 2);
  float*  logits = (float*)take((size_t)KBEAM * VOCAB * 4);
  float*  tmp    = (float*)take(H * 4);
  int*    tokens = (int*)take(KBEAM * 4);
  float*  logp   = (float*)take(KBEAM * 4);
  int*    lens   = (int*)take(KBEAM * 4);
  int*    fin    = (int*)take(KBEAM * 4);
  int*    seqs   = (int*)take(KBEAM * TMAX * 4);

  dim3 tb(32, 8);
  transpose_f32_kernel<<<dim3(H3 / 32, H / 32), tb, 0, stream>>>(Wx_e, WxeT, H, H3);
  transpose_f32_kernel<<<dim3(H3 / 32, H / 32), tb, 0, stream>>>(Wh_e, WheT, H, H3);
  transpose_f32_kernel<<<dim3(H3 / 32, H2 / 32), tb, 0, stream>>>(Wx_d, WxdT, H2, H3);
  transpose_f32_kernel<<<dim3(H3 / 32, H / 32), tb, 0, stream>>>(Wh_d, WhdT, H, H3);
  transpose_f32_kernel<<<dim3(H / 32, H / 32), tb, 0, stream>>>(W_encl, WenclT, H, H);
  transpose_f32_kernel<<<dim3(H / 32, H / 32), tb, 0, stream>>>(W_lin, WlinT, H, H);
  transpose_f32_bf16_kernel<<<dim3(VOCAB / 32, H2 / 32), tb, 0, stream>>>(W_out, WoutT, H2, VOCAB);

  init_kernel<<<64, 512, 0, stream>>>(trg, tokens, henc_a, xo);

  // encoder: 512 sequential GRU steps (weights L2-resident after step 1)
  for (int t = 0; t < SLEN; t++) {
    const float* hin = (t & 1) ? henc_b : henc_a;
    float* hout = (t & 1) ? henc_a : henc_b;
    enc_step_kernel<<<128, 256, 0, stream>>>(src, emb_src, WxeT, WheT, b_e,
                                             hin, hout, enc_out + (size_t)t * H, t);
  }
  // SLEN even -> final h in henc_a
  linear_tanh_kernel<<<128, 256, 0, stream>>>(WenclT, b_encl, henc_a, tmp);
  linear_tanh_kernel<<<128, 256, 0, stream>>>(WlinT, b_lin, tmp, hidA);  // hidden0 -> beam row 0

  // greedy seed step (B=1)
  dec_attn_kernel<<<1, 1024, 0, stream>>>(enc_out, hidA, tokens, emb_tgt, xg);
  dec_gru_kernel<<<128, 256, 0, stream>>>(WxdT, WhdT, b_d, xg, hidA, hidB, xo);
  dec_logits_kernel<<<VOCAB / 16 / 8, 256, 0, stream>>>(xo, WoutT, b_out, logits);
  greedy_pick_kernel<<<1, 1024, 0, stream>>>(logits, tokens, logp, lens, fin, seqs, hidB, hidA);

  // 63 beam steps (B=4)
  for (int t = 0; t < TMAX - 1; t++) {
    dec_attn_kernel<<<KBEAM, 1024, 0, stream>>>(enc_out, hidA, tokens, emb_tgt, xg);
    dec_gru_kernel<<<KBEAM * 128, 256, 0, stream>>>(WxdT, WhdT, b_d, xg, hidA, hidB, xo);
    dec_logits_kernel<<<VOCAB / 16 / 8, 256, 0, stream>>>(xo, WoutT, b_out, logits);
    beam_topk_kernel<<<1, 1024, 0, stream>>>(logits, tokens, logp, lens, fin, seqs, hidB, hidA, t);
  }
  finalize_kernel<<<1, 512, 0, stream>>>(seqs, logp, lens, (float*)d_out);
}
